// LearnableMultiHeadSelfAttention_55224689492518
// MI455X (gfx1250) — matrix operand order repaired and checked by reading, not compiled, not run on hardware
//
#include <hip/hip_runtime.h>
#include <hip/hip_bf16.h>

typedef __bf16 bf16_t;
typedef __attribute__((ext_vector_type(16))) __bf16 v16bf;
typedef __attribute__((ext_vector_type(8)))  float  v8f;

#define B_DIM 4
#define I_DIM 512
#define H_DIM 1024
#define NH    16
#define D_DIM 64
#define N_SEG 4
#define L_SEG 512          // log2 = 9
#define CLEN  2048         // N_SEG * L_SEG
#define R_DIM 2560         // CLEN + I_DIM
#define LTOT  2560

union FragU { v16bf v; unsigned int u[8]; bf16_t h[16]; };

// ---------------------------------------------------------------------------
// Cast kernels
// ---------------------------------------------------------------------------
__global__ void cast_f32_bf16(const float* __restrict__ in, bf16_t* __restrict__ out, size_t n) {
    size_t i = (size_t)blockIdx.x * blockDim.x + threadIdx.x;
    if (i < n) out[i] = (bf16_t)in[i];
}

// cache (N,B,L,H) f32 -> cfb (B, N*L, H) bf16
__global__ void cast_cache_bf16(const float* __restrict__ in, bf16_t* __restrict__ out) {
    size_t i = (size_t)blockIdx.x * blockDim.x + threadIdx.x;
    int h   = (int)(i & (H_DIM - 1));
    size_t row = i >> 10;                 // H = 1024
    int b   = (int)(row / CLEN);
    int nj  = (int)(row % CLEN);
    int n   = nj >> 9;                    // / L_SEG
    int l   = nj & (L_SEG - 1);
    out[i] = (bf16_t)in[(((size_t)n * B_DIM + b) * L_SEG + l) * H_DIM + h];
}

// pos_emb (R,B,H) f32 -> peb (B,R,H) bf16
__global__ void cast_pos_bf16(const float* __restrict__ in, bf16_t* __restrict__ out) {
    size_t i = (size_t)blockIdx.x * blockDim.x + threadIdx.x;
    int h   = (int)(i & (H_DIM - 1));
    size_t row = i >> 10;
    int b   = (int)(row / R_DIM);
    int t   = (int)(row % R_DIM);
    out[i] = (bf16_t)in[((size_t)t * B_DIM + b) * H_DIM + h];
}

// ---------------------------------------------------------------------------
// Generic bf16 WMMA GEMM:  C[m,o] = sum_k A[m,k] * W[o,k] + bias[o]
// Per wave: 16 rows x 64 cols.  Block: 8 waves -> 128 x 64 tile.
// Register double-buffered over the K loop so fragment loads for step k+1
// overlap the 4 WMMAs of step k.
// mode: 0 = bf16 head layout out1          ((b*NH+n)*P + i)*D + d
//       1 = dual qu/qv head layout         t=(acc+bias)*scale; out1=t+ub, out2=t+vb
//       2 = f32 row-major out_f32          (final projection)
//       3 = bf16 TRANSPOSED head layout    ((b*NH+n)*D + d)*P + i   (for V/cacheV)
// ---------------------------------------------------------------------------
__global__ __launch_bounds__(256) void gemm_bf16_kernel(
    const bf16_t* __restrict__ A, const bf16_t* __restrict__ W,
    const float* __restrict__ bias, int M, int P, int mode,
    bf16_t* __restrict__ out1, bf16_t* __restrict__ out2,
    const float* __restrict__ ub, const float* __restrict__ vb, float scale,
    float* __restrict__ out_f32)
{
    const int lane = threadIdx.x & 31;
    const int wave = threadIdx.x >> 5;
    const int m0 = blockIdx.x * 128 + wave * 16;
    const int n0 = blockIdx.y * 64;
    if (m0 >= M) return;

    const int kofsA = (lane >> 4) * 8;
    const int kofsB = (lane >> 4) * 16;
    const int mloc  = lane & 15;
    const bf16_t* Arow = A + (size_t)(m0 + mloc) * H_DIM;
    const bf16_t* Wbase = W + (size_t)(n0 + mloc) * H_DIM;

    v8f acc[4] = {};
    FragU fa[2];
    FragU fb[2][4];

    auto load_step = [&](int k0, int buf) {
#pragma unroll
        for (int r = 0; r < 8; ++r) {
            int k = k0 + 2 * r + kofsB;
            fa[buf].u[r] = *(const unsigned int*)(Arow + k);
        }
#pragma unroll
        for (int c = 0; c < 4; ++c) {
            const bf16_t* Wrow = Wbase + (size_t)c * 16 * H_DIM + k0 + kofsB;
#pragma unroll
            for (int r = 0; r < 8; ++r)
                fb[buf][c].u[r] = *(const unsigned int*)(Wrow + 2 * r);
        }
    };

    load_step(0, 0);
#pragma unroll 4
    for (int k0 = 0; k0 < H_DIM; k0 += 32) {
        const int cur = (k0 >> 5) & 1;
        if (k0 + 32 < H_DIM) {
            // prefetch one step further ahead into L2 (global_prefetch_b8)
            if (k0 + 64 < H_DIM) {
                __builtin_prefetch(Arow + k0 + 64, 0, 0);
                __builtin_prefetch(Wbase + k0 + 64, 0, 0);
            }
            load_step(k0 + 32, cur ^ 1);
        }
#pragma unroll
        for (int c = 0; c < 4; ++c)
            acc[c] = __builtin_amdgcn_wmma_f32_16x16x32_bf16(
                false, fa[cur].v, false, fb[cur][c].v, (short)0, acc[c], false, false);
    }

    const int rbase = (lane >> 4) * 8;
    const int ccol  = lane & 15;
    const int b     = m0 / P;
    const int iloc0 = m0 % P;
#pragma unroll
    for (int c = 0; c < 4; ++c) {
#pragma unroll
        for (int e = 0; e < 8; ++e) {
            int rrow = rbase + e;
            int o = n0 + c * 16 + ccol;
            float v = acc[c][e] + bias[o];
            int n = o >> 6, d = o & 63;
            if (mode == 2) {
                out_f32[(size_t)(m0 + rrow) * H_DIM + o] = v;
            } else if (mode == 1) {
                size_t dst = (((size_t)(b * NH + n) * P) + (iloc0 + rrow)) * D_DIM + d;
                float t = v * scale;
                out1[dst] = (bf16_t)(t + ub[o]);
                out2[dst] = (bf16_t)(t + vb[o]);
            } else if (mode == 3) {
                size_t dst = (((size_t)(b * NH + n) * D_DIM) + d) * P + iloc0 + rrow;
                out1[dst] = (bf16_t)v;
            } else {
                size_t dst = (((size_t)(b * NH + n) * P) + (iloc0 + rrow)) * D_DIM + d;
                out1[dst] = (bf16_t)v;
            }
        }
    }
}

// ---------------------------------------------------------------------------
// Fused relative attention with segment cache, flash-style online softmax.
// One wave per 16-row query tile; block = 4 waves.
// V and cache-V are consumed in TRANSPOSED head layout (K-major), so all
// WMMA B fragments load as contiguous 32-bit pairs.
// ---------------------------------------------------------------------------
__global__ __launch_bounds__(128) void attn_kernel(
    const bf16_t* __restrict__ QU, const bf16_t* __restrict__ QV,
    const bf16_t* __restrict__ KH, const bf16_t* __restrict__ VT,
    const bf16_t* __restrict__ RH, const bf16_t* __restrict__ CK,
    const bf16_t* __restrict__ CVT,
    const float* __restrict__ bias, const float* __restrict__ indice,
    const float* __restrict__ sentw,
    bf16_t* __restrict__ OHb)
{
    __shared__ __align__(16) float  bds_s[4][16 * 48];
    __shared__ __align__(16) bf16_t ps_s[4][16 * 32];

    const int lane = threadIdx.x & 31;
    const int wave = threadIdx.x >> 5;
    const int tile = blockIdx.x * 4 + wave;
    const int i0 = (tile & (I_DIM / 16 - 1)) * 16;
    const int n  = (tile / (I_DIM / 16)) & (NH - 1);
    const int b  = tile / ((I_DIM / 16) * NH);

    const int rbase = (lane >> 4) * 8;
    const int ccol  = lane & 15;
    const int kofsA = (lane >> 4) * 8;
    const int kofsB = (lane >> 4) * 16;

    // --- load Q fragments (qu, qv), two K-chunks of 32 over D=64 ---
    v16bf qua[2], qva[2];
    {
        const bf16_t* pu = QU + (((size_t)(b * NH + n) * I_DIM) + i0 + (lane & 15)) * D_DIM;
        const bf16_t* pv = QV + (((size_t)(b * NH + n) * I_DIM) + i0 + (lane & 15)) * D_DIM;
#pragma unroll
        for (int c = 0; c < 2; ++c) {
            FragU fu, fv;
#pragma unroll
            for (int r = 0; r < 8; ++r) {
                int k = c * 32 + 2 * r + kofsB;
                fu.u[r] = *(const unsigned int*)(pu + k);
                fv.u[r] = *(const unsigned int*)(pv + k);
            }
            qua[c] = fu.v; qva[c] = fv.v;
        }
    }

    const bf16_t* Khn   = KH  + (size_t)(b * NH + n) * I_DIM * D_DIM;
    const bf16_t* VThn  = VT  + (size_t)(b * NH + n) * D_DIM * I_DIM;
    const bf16_t* Rhn   = RH  + (size_t)(b * NH + n) * R_DIM * D_DIM;
    const bf16_t* CKhn  = CK  + (size_t)(b * NH + n) * CLEN * D_DIM;
    const bf16_t* CVThn = CVT + (size_t)(b * NH + n) * D_DIM * CLEN;

    float mrun[8], srun[8];
#pragma unroll
    for (int e = 0; e < 8; ++e) { mrun[e] = -1e30f; srun[e] = 0.f; }
    v8f Oacc[4] = {};

    float* bds = bds_s[wave];
    bf16_t* ps = ps_s[wave];

    for (int j0 = 0; j0 < LTOT; j0 += 32) {
        // prefetch next block's K row for this lane (global_prefetch_b8)
        {
            int jn = j0 + 32 + ccol;
            if (jn < LTOT) {
                const bf16_t* pn = (jn < CLEN) ? (CKhn + (size_t)jn * D_DIM)
                                               : (Khn + (size_t)(jn - CLEN) * D_DIM);
                __builtin_prefetch(pn, 0, 0);
            }
        }
        const bool incache = (j0 < CLEN);       // whole 32-block on one side
        const int  seg     = j0 >> 9;           // L segment (valid when incache)

        // ---------- scores: two 16x16 tiles ----------
        v8f S[2];
#pragma unroll
        for (int jt = 0; jt < 2; ++jt) {
            int j = j0 + jt * 16 + ccol;
            const bf16_t* krow = incache ? (CKhn + (size_t)j * D_DIM)
                                         : (Khn + (size_t)(j - CLEN) * D_DIM);
            v8f a = {};
#pragma unroll
            for (int c = 0; c < 2; ++c) {
                FragU fb;
#pragma unroll
                for (int r = 0; r < 8; ++r)
                    fb.u[r] = *(const unsigned int*)(krow + c * 32 + kofsB + 2 * r);
                a = __builtin_amdgcn_wmma_f32_16x16x32_bf16(
                    false, qua[c], false, fb.v, (short)0, a, false, false);
            }
            S[jt] = a;
        }
        // AC epilogue: indice_bool scale on cache cols, bias on self cols
#pragma unroll
        for (int e = 0; e < 8; ++e) {
            int i = i0 + rbase + e;
            if (incache) {
                float ind = indice[((size_t)i * B_DIM + b) * N_SEG + seg];
                S[0][e] *= ind;
                S[1][e] *= ind;
            } else {
                S[0][e] += bias[(size_t)i * I_DIM + (j0 + ccol - CLEN)];
                S[1][e] += bias[(size_t)i * I_DIM + (j0 + 16 + ccol - CLEN)];
            }
        }

        // ---------- BD band: 16 x 48 of qv . RH, then rel-shift gather ----------
        const int cb0 = j0 + I_DIM - 16 - i0;   // >= 0 always
#pragma unroll
        for (int t = 0; t < 3; ++t) {
            int bc = cb0 + t * 16 + ccol;
            bool oob = (bc >= LTOT);
            const bf16_t* rrow = Rhn + (size_t)(oob ? 0 : bc) * D_DIM;
            v8f a = {};
#pragma unroll
            for (int c = 0; c < 2; ++c) {
                FragU fb;
#pragma unroll
                for (int r = 0; r < 8; ++r)
                    fb.u[r] = oob ? 0u : *(const unsigned int*)(rrow + c * 32 + kofsB + 2 * r);
                a = __builtin_amdgcn_wmma_f32_16x16x32_bf16(
                    false, qva[c], false, fb.v, (short)0, a, false, false);
            }
#pragma unroll
            for (int e = 0; e < 8; ++e)
                bds[(rbase + e) * 48 + t * 16 + ccol] = a[e];
        }
        asm volatile("s_wait_dscnt 0" ::: "memory");
#pragma unroll
        for (int jt = 0; jt < 2; ++jt) {
#pragma unroll
            for (int e = 0; e < 8; ++e) {
                int rrow = rbase + e;
                int jjl = jt * 16 + ccol + 15 - rrow;     // 0..46
                int jj = cb0 + jjl;
                if (jj < LTOT) {
                    float bd = bds[rrow * 48 + jjl];
                    if (jj < CLEN)
                        bd *= indice[((size_t)(i0 + rrow) * B_DIM + b) * N_SEG + (jj >> 9)];
                    S[jt][e] += bd;
                }
            }
        }

        // ---------- online softmax over these 32 columns ----------
#pragma unroll
        for (int e = 0; e < 8; ++e) {
            float v0 = S[0][e], v1 = S[1][e];
            float mx = fmaxf(v0, v1);
#pragma unroll
            for (int d = 8; d >= 1; d >>= 1) mx = fmaxf(mx, __shfl_xor(mx, d, 16));
            float mnew = fmaxf(mrun[e], mx);
            float f = __expf(mrun[e] - mnew);
            float p0 = __expf(v0 - mnew), p1 = __expf(v1 - mnew);
            float rs = p0 + p1;
#pragma unroll
            for (int d = 8; d >= 1; d >>= 1) rs += __shfl_xor(rs, d, 16);
            srun[e] = srun[e] * f + rs;
            mrun[e] = mnew;
#pragma unroll
            for (int c = 0; c < 4; ++c) Oacc[c][e] *= f;

            // stage P with sent_weights folded into cache columns
            if (incache) {
                float sw = sentw[((size_t)(i0 + rbase + e) * B_DIM + b) * N_SEG + seg];
                p0 *= sw; p1 *= sw;
            }
            ps[(rbase + e) * 32 + ccol]      = (bf16_t)p0;
            ps[(rbase + e) * 32 + 16 + ccol] = (bf16_t)p1;
        }
        asm volatile("s_wait_dscnt 0" ::: "memory");

        // ---------- P (16x32) @ V (32x64), V in K-major transposed layout ----------
        FragU pf;
        {
            const bf16_t* pp = ps + (lane & 15) * 32;
#pragma unroll
            for (int r = 0; r < 8; ++r) {
                int k = 2 * r + kofsB;
                pf.u[r] = *(const unsigned int*)(pp + k);
            }
        }
#pragma unroll
        for (int c = 0; c < 4; ++c) {
            int d = c * 16 + ccol;
            const bf16_t* vrow = incache
                ? (CVThn + (size_t)d * CLEN + j0)
                : (VThn + (size_t)d * I_DIM + (j0 - CLEN));
            FragU fv;
#pragma unroll
            for (int r = 0; r < 8; ++r)
                fv.u[r] = *(const unsigned int*)(vrow + kofsB + 2 * r);
            Oacc[c] = __builtin_amdgcn_wmma_f32_16x16x32_bf16(
                false, pf.v, false, fv.v, (short)0, Oacc[c], false, false);
        }
    }

    // ---------- finalize: divide by softmax sum, store (b,i,h) bf16 ----------
#pragma unroll
    for (int c = 0; c < 4; ++c) {
#pragma unroll
        for (int e = 0; e < 8; ++e) {
            int i = i0 + rbase + e;
            int h = n * D_DIM + c * 16 + ccol;
            OHb[((size_t)b * I_DIM + i) * H_DIM + h] = (bf16_t)(Oacc[c][e] / srun[e]);
        }
    }
}

// ---------------------------------------------------------------------------
extern "C" void kernel_launch(void* const* d_in, const int* in_sizes, int n_in,
                              void* d_out, int out_size, void* d_ws, size_t ws_size,
                              hipStream_t stream) {
    const float* x       = (const float*)d_in[0];
    const float* bias    = (const float*)d_in[1];
    const float* pos_emb = (const float*)d_in[2];
    const float* pbu     = (const float*)d_in[3];
    const float* pbv     = (const float*)d_in[4];
    const float* cache   = (const float*)d_in[5];
    const float* indice  = (const float*)d_in[6];
    const float* sentw   = (const float*)d_in[7];
    const float* Wq = (const float*)d_in[8];  const float* bq = (const float*)d_in[9];
    const float* Wk = (const float*)d_in[10]; const float* bk = (const float*)d_in[11];
    const float* Wv = (const float*)d_in[12]; const float* bv = (const float*)d_in[13];
    const float* Wo = (const float*)d_in[14]; const float* bo = (const float*)d_in[15];
    const float* Wr = (const float*)d_in[16]; const float* br = (const float*)d_in[17];

    char* ws = (char*)d_ws;
    size_t off = 0;
    auto take = [&](size_t elems) -> bf16_t* {
        bf16_t* p = (bf16_t*)(ws + off);
        off = (off + elems * sizeof(bf16_t) + 255) & ~(size_t)255;
        return p;
    };
    const size_t SZ_XB  = (size_t)B_DIM * I_DIM * H_DIM;
    const size_t SZ_CFB = (size_t)B_DIM * CLEN * H_DIM;
    const size_t SZ_PEB = (size_t)B_DIM * R_DIM * H_DIM;
    const size_t SZ_W   = (size_t)H_DIM * H_DIM;
    const size_t SZ_QH  = (size_t)B_DIM * NH * I_DIM * D_DIM;
    const size_t SZ_RH  = (size_t)B_DIM * NH * R_DIM * D_DIM;
    const size_t SZ_CH  = (size_t)B_DIM * NH * CLEN * D_DIM;

    bf16_t* xb  = take(SZ_XB);
    bf16_t* cfb = take(SZ_CFB);
    bf16_t* peb = take(SZ_PEB);
    bf16_t* Wqb = take(SZ_W);
    bf16_t* Wkb = take(SZ_W);
    bf16_t* Wvb = take(SZ_W);
    bf16_t* Wrb = take(SZ_W);
    bf16_t* Wob = take(SZ_W);
    bf16_t* QU  = take(SZ_QH);
    bf16_t* QV  = take(SZ_QH);
    bf16_t* KH  = take(SZ_QH);
    bf16_t* VT  = take(SZ_QH);   // transposed head layout
    bf16_t* RHp = take(SZ_RH);
    bf16_t* CK  = take(SZ_CH);
    bf16_t* CVT = take(SZ_CH);   // transposed head layout
    bf16_t* OHb = take(SZ_XB);

    // casts
    cast_f32_bf16<<<dim3((unsigned)(SZ_XB / 256)), 256, 0, stream>>>(x, xb, SZ_XB);
    cast_cache_bf16<<<dim3((unsigned)(SZ_CFB / 256)), 256, 0, stream>>>(cache, cfb);
    cast_pos_bf16<<<dim3((unsigned)(SZ_PEB / 256)), 256, 0, stream>>>(pos_emb, peb);
    cast_f32_bf16<<<dim3((unsigned)(SZ_W / 256)), 256, 0, stream>>>(Wq, Wqb, SZ_W);
    cast_f32_bf16<<<dim3((unsigned)(SZ_W / 256)), 256, 0, stream>>>(Wk, Wkb, SZ_W);
    cast_f32_bf16<<<dim3((unsigned)(SZ_W / 256)), 256, 0, stream>>>(Wv, Wvb, SZ_W);
    cast_f32_bf16<<<dim3((unsigned)(SZ_W / 256)), 256, 0, stream>>>(Wr, Wrb, SZ_W);
    cast_f32_bf16<<<dim3((unsigned)(SZ_W / 256)), 256, 0, stream>>>(Wo, Wob, SZ_W);

    const float qscale = 0.125f;  // D^-0.5, D=64
    dim3 blk(256);
    // q -> qu/qv (dual epilogue, mode 1)
    gemm_bf16_kernel<<<dim3(B_DIM * I_DIM / 128, H_DIM / 64), blk, 0, stream>>>(
        xb, Wqb, bq, B_DIM * I_DIM, I_DIM, 1, QU, QV, pbu, pbv, qscale, nullptr);
    // k (head layout, mode 0), v (transposed head layout, mode 3)
    gemm_bf16_kernel<<<dim3(B_DIM * I_DIM / 128, H_DIM / 64), blk, 0, stream>>>(
        xb, Wkb, bk, B_DIM * I_DIM, I_DIM, 0, KH, nullptr, nullptr, nullptr, 1.f, nullptr);
    gemm_bf16_kernel<<<dim3(B_DIM * I_DIM / 128, H_DIM / 64), blk, 0, stream>>>(
        xb, Wvb, bv, B_DIM * I_DIM, I_DIM, 3, VT, nullptr, nullptr, nullptr, 1.f, nullptr);
    // cache k (mode 0), cache v (mode 3)
    gemm_bf16_kernel<<<dim3(B_DIM * CLEN / 128, H_DIM / 64), blk, 0, stream>>>(
        cfb, Wkb, bk, B_DIM * CLEN, CLEN, 0, CK, nullptr, nullptr, nullptr, 1.f, nullptr);
    gemm_bf16_kernel<<<dim3(B_DIM * CLEN / 128, H_DIM / 64), blk, 0, stream>>>(
        cfb, Wvb, bv, B_DIM * CLEN, CLEN, 3, CVT, nullptr, nullptr, nullptr, 1.f, nullptr);
    // r (mode 0)
    gemm_bf16_kernel<<<dim3(B_DIM * R_DIM / 128, H_DIM / 64), blk, 0, stream>>>(
        peb, Wrb, br, B_DIM * R_DIM, R_DIM, 0, RHp, nullptr, nullptr, nullptr, 1.f, nullptr);

    // fused attention
    attn_kernel<<<dim3(B_DIM * NH * (I_DIM / 16) / 4), dim3(128), 0, stream>>>(
        QU, QV, KH, VT, RHp, CK, CVT, bias, indice, sentw, OHb);

    // output projection -> f32 d_out (mode 2)
    gemm_bf16_kernel<<<dim3(B_DIM * I_DIM / 128, H_DIM / 64), blk, 0, stream>>>(
        OHb, Wob, bo, B_DIM * I_DIM, I_DIM, 2, nullptr, nullptr, nullptr, nullptr, 1.f,
        (float*)d_out);

    (void)in_sizes; (void)n_in; (void)out_size; (void)ws_size;
}
